// Embedding2Score_2259152798068
// MI455X (gfx1250) — matrix operand order repaired and checked by reading, not compiled, not run on hardware
//
#include <hip/hip_runtime.h>
#include <hip/hip_bf16.h>

typedef __attribute__((ext_vector_type(16))) __bf16 v16bf;
typedef __attribute__((ext_vector_type(8)))  float  v8f;

#define N_NODES 131072
#define B_SESS  4096
#define D_HID   128
#define V_VOCAB 50000
#define LDB_P   136   // 128 + 8 bf16 padding -> 272B row stride, conflict-free

// ---- WMMA fragment loaders (ISA 7.12.2 layouts, wave32) -------------------
// A: 16x32 bf16. lane = (r, half). elems 0..7 = K[k0+half*8 .. +7],
//                                  elems 8..15 = K[k0+16+half*8 .. +7]
static __device__ __forceinline__ v16bf load_a16(const __bf16* base, int ld,
                                                 int m0, int k0, int r, int half) {
  const __bf16* p = base + (size_t)(m0 + r) * ld + k0 + half * 8;
  v16bf a;
  ((float4*)&a)[0] = *(const float4*)(p);
  ((float4*)&a)[1] = *(const float4*)(p + 16);
  return a;
}
// B: 32x16 bf16 with B[k][n] = Bmat[n][k] (Bmat row-major K-contiguous).
// lane (r, half): elems 0..7 = Bmat[n0+r][k0+half*8 .. +7], elems 8..15 = Bmat[n0+r][k0+16+half*8 .. +7].
static __device__ __forceinline__ v16bf load_b16(const __bf16* base, int ld,
                                                 int n0, int k0, int r, int half) {
  const __bf16* p = base + (size_t)(n0 + r) * ld + k0 + half * 8;
  v16bf b;
  ((float4*)&b)[0] = *(const float4*)(p);
  ((float4*)&b)[1] = *(const float4*)(p + 16);
  return b;
}

// ---- small helper kernels --------------------------------------------------
__global__ void k_cvt_bf16(const float* __restrict__ src, __bf16* __restrict__ dst, int n) {
  int i = blockIdx.x * blockDim.x + threadIdx.x;
  int stride = gridDim.x * blockDim.x;
  for (; i < n; i += stride) dst[i] = (__bf16)src[i];
}

__global__ void k_find_last(const int* __restrict__ batch, int* __restrict__ last_idx, int n) {
  int i = blockIdx.x * blockDim.x + threadIdx.x;
  if (i < n) {
    int b = batch[i];
    if (i == n - 1 || batch[i + 1] != b) last_idx[b] = i;
  }
}

// 8 sessions per block: gather v_n and q1s[b] = v_n @ W1^T + b1
__global__ void k_vn_q1(const float* __restrict__ X, const int* __restrict__ last_idx,
                        const float* __restrict__ W1w, const float* __restrict__ W1b,
                        float* __restrict__ v_n, float* __restrict__ q1s) {
  __shared__ float vn_s[8][D_HID];
  int d  = threadIdx.x;
  int b0 = blockIdx.x * 8;
  for (int s = 0; s < 8; ++s) {
    int li = last_idx[b0 + s];
    float v = X[(size_t)li * D_HID + d];
    vn_s[s][d] = v;
    v_n[(size_t)(b0 + s) * D_HID + d] = v;
  }
  __syncthreads();
  float acc[8];
  float bias = W1b[d];
#pragma unroll
  for (int s = 0; s < 8; ++s) acc[s] = bias;
  const float* wrow = W1w + (size_t)d * D_HID;
#pragma unroll 4
  for (int k = 0; k < D_HID; ++k) {
    float w = wrow[k];
#pragma unroll
    for (int s = 0; s < 8; ++s) acc[s] += w * vn_s[s][k];
  }
  for (int s = 0; s < 8; ++s) q1s[(size_t)(b0 + s) * D_HID + d] = acc[s];
}

// Fused attention: per wave, 16-node tile. q2 via WMMA, then
// alpha[i] = q_w . sigmoid(q1s[batch[i]] + q2_i + b2) + q_b
__global__ void k_alpha(const __bf16* __restrict__ Xb, const __bf16* __restrict__ W2b16,
                        const float* __restrict__ W2bias, const float* __restrict__ q1s,
                        const int* __restrict__ batch, const float* __restrict__ qw,
                        const float* __restrict__ qb, float* __restrict__ alpha) {
  int lane = threadIdx.x & 31;
  int wave = threadIdx.x >> 5;
  int r = lane & 15, half = lane >> 4;
  int tile = blockIdx.x * 8 + wave;   // N/16 = 8192 tiles
  int i0 = tile * 16;

  v16bf a[4];
#pragma unroll
  for (int kc = 0; kc < 4; ++kc) a[kc] = load_a16(Xb, D_HID, i0, kc * 32, r, half);

  const float* qp[8];
#pragma unroll
  for (int v = 0; v < 8; ++v)
    qp[v] = q1s + (size_t)batch[i0 + half * 8 + v] * D_HID;

  float part[8] = {0.f, 0.f, 0.f, 0.f, 0.f, 0.f, 0.f, 0.f};
#pragma unroll 2
  for (int nc = 0; nc < 8; ++nc) {
    v8f c = {0.f, 0.f, 0.f, 0.f, 0.f, 0.f, 0.f, 0.f};
#pragma unroll
    for (int kc = 0; kc < 4; ++kc) {
      v16bf bb = load_b16(W2b16, D_HID, nc * 16, kc * 32, r, half);
      c = __builtin_amdgcn_wmma_f32_16x16x32_bf16(false, a[kc], false, bb,
                                                  (short)0, c, false, false);
    }
    int n = nc * 16 + r;
    float wq = qw[n];
    float wb = W2bias[n];
#pragma unroll
    for (int v = 0; v < 8; ++v) {
      float t = c[v] + qp[v][n] + wb;
      float sg = 1.0f / (1.0f + __expf(-t));
      part[v] += sg * wq;
    }
  }
  float qb0 = qb[0];
#pragma unroll
  for (int v = 0; v < 8; ++v) {
    float s = part[v];
    s += __shfl_xor(s, 1, 32);
    s += __shfl_xor(s, 2, 32);
    s += __shfl_xor(s, 4, 32);
    s += __shfl_xor(s, 8, 32);
    if (r == 0) alpha[i0 + half * 8 + v] = s + qb0;
  }
}

// Per-session: s_g = sum alpha_i * x_i (batch is sorted -> contiguous range),
// then s_h = [v_n, s_g] @ W3^T + b3, stored as bf16 for the final GEMM.
__global__ void k_sg_sh(const float* __restrict__ X, const float* __restrict__ alpha,
                        const int* __restrict__ last_idx, const float* __restrict__ v_n,
                        const float* __restrict__ W3w, const float* __restrict__ W3b,
                        __bf16* __restrict__ shb) {
  __shared__ float sg[D_HID];
  __shared__ float vn[D_HID];
  int d = threadIdx.x;
  int b = blockIdx.x;
  int start = (b == 0) ? 0 : (last_idx[b - 1] + 1);
  int end = last_idx[b];
  float acc = 0.f;
  for (int i = start; i <= end; ++i) acc += alpha[i] * X[(size_t)i * D_HID + d];
  sg[d] = acc;
  vn[d] = v_n[(size_t)b * D_HID + d];
  __syncthreads();
  float sh = W3b[d];
  const float* wr = W3w + (size_t)d * 2 * D_HID;
#pragma unroll 4
  for (int k = 0; k < D_HID; ++k) sh += wr[k] * vn[k] + wr[D_HID + k] * sg[k];
  shb[(size_t)b * D_HID + d] = (__bf16)sh;
}

// Final scores: out[4096,50000] = s_h @ item_weight^T.
// Block = 256M x 64N, 8 waves x (32M x 64N) each. item_weight tile is staged
// into LDS with CDNA5 async global->LDS copies (ASYNCcnt), f32 accumulate.
__global__ void __launch_bounds__(256)
k_scores(const __bf16* __restrict__ shb, const __bf16* __restrict__ itemb,
         float* __restrict__ out) {
  __shared__ __attribute__((aligned(16))) __bf16 Bs[64 * LDB_P];
  int nBase = blockIdx.x * 64;
  int mBase = blockIdx.y * 256;

  // ---- async stage of B tile: 64 rows x 128 K bf16 = 16 KB ----
  {
    int t = threadIdx.x;
    int row = t >> 2;
    int kseg = (t & 3) * 32;                      // 32 bf16 = 64 B per thread
    int srow = nBase + row;
    if (srow > V_VOCAB - 1) srow = V_VOCAB - 1;   // clamp tail, loads stay in-bounds
    const __bf16* src = itemb + (size_t)srow * D_HID + kseg;
    unsigned ldsa = (unsigned)(unsigned long long)(&Bs[row * LDB_P + kseg]);
#pragma unroll
    for (int q = 0; q < 4; ++q) {
      unsigned long long ga = (unsigned long long)(src + q * 8);
      unsigned la = ldsa + q * 16;
      asm volatile("global_load_async_to_lds_b128 %0, %1, off"
                   :: "v"(la), "v"(ga) : "memory");
    }
  }
  asm volatile("s_wait_asynccnt 0" ::: "memory");
  __syncthreads();

  int lane = threadIdx.x & 31;
  int wave = threadIdx.x >> 5;
  int r = lane & 15, half = lane >> 4;
  int m0 = mBase + wave * 32;                     // two 16-row subtiles per wave

  v16bf a0[4], a1[4];
#pragma unroll
  for (int kc = 0; kc < 4; ++kc) {
    a0[kc] = load_a16(shb, D_HID, m0,      kc * 32, r, half);
    a1[kc] = load_a16(shb, D_HID, m0 + 16, kc * 32, r, half);
  }

#pragma unroll
  for (int nt = 0; nt < 4; ++nt) {
    v8f c0 = {0.f, 0.f, 0.f, 0.f, 0.f, 0.f, 0.f, 0.f};
    v8f c1 = {0.f, 0.f, 0.f, 0.f, 0.f, 0.f, 0.f, 0.f};
#pragma unroll
    for (int kc = 0; kc < 4; ++kc) {
      v16bf bb = load_b16(Bs, LDB_P, nt * 16, kc * 32, r, half);
      c0 = __builtin_amdgcn_wmma_f32_16x16x32_bf16(false, a0[kc], false, bb,
                                                   (short)0, c0, false, false);
      c1 = __builtin_amdgcn_wmma_f32_16x16x32_bf16(false, a1[kc], false, bb,
                                                   (short)0, c1, false, false);
    }
    int n = nBase + nt * 16 + r;
    if (n < V_VOCAB) {
#pragma unroll
      for (int v = 0; v < 8; ++v) {
        out[(size_t)(m0 +      half * 8 + v) * V_VOCAB + n] = c0[v];
        out[(size_t)(m0 + 16 + half * 8 + v) * V_VOCAB + n] = c1[v];
      }
    }
  }
}

// ---- launch ----------------------------------------------------------------
extern "C" void kernel_launch(void* const* d_in, const int* in_sizes, int n_in,
                              void* d_out, int out_size, void* d_ws, size_t ws_size,
                              hipStream_t stream) {
  const float* X     = (const float*)d_in[0];
  const float* itemW = (const float*)d_in[1];
  const float* W1w   = (const float*)d_in[2];
  const float* W1b   = (const float*)d_in[3];
  const float* W2w   = (const float*)d_in[4];
  const float* W2b   = (const float*)d_in[5];
  const float* qw    = (const float*)d_in[6];
  const float* qb    = (const float*)d_in[7];
  const float* W3w   = (const float*)d_in[8];
  const float* W3b   = (const float*)d_in[9];
  const int*   batch = (const int*)d_in[10];
  float* out = (float*)d_out;

  char* ws = (char*)d_ws;
  size_t off = 0;
  auto alloc = [&](size_t bytes) -> void* {
    void* p = ws + off;
    off = (off + bytes + 255) & ~(size_t)255;
    return p;
  };
  __bf16* Xb    = (__bf16*)alloc((size_t)N_NODES * D_HID * sizeof(__bf16));
  __bf16* itemb = (__bf16*)alloc((size_t)V_VOCAB * D_HID * sizeof(__bf16));
  __bf16* W2b16 = (__bf16*)alloc((size_t)D_HID * D_HID * sizeof(__bf16));
  int*    lastI = (int*)alloc((size_t)B_SESS * sizeof(int));
  float*  v_n   = (float*)alloc((size_t)B_SESS * D_HID * sizeof(float));
  float*  q1s   = (float*)alloc((size_t)B_SESS * D_HID * sizeof(float));
  float*  alpha = (float*)alloc((size_t)N_NODES * sizeof(float));
  __bf16* shb   = (__bf16*)alloc((size_t)B_SESS * D_HID * sizeof(__bf16));

  k_cvt_bf16<<<8192, 256, 0, stream>>>(X, Xb, N_NODES * D_HID);
  k_cvt_bf16<<<8192, 256, 0, stream>>>(itemW, itemb, V_VOCAB * D_HID);
  k_cvt_bf16<<<64, 256, 0, stream>>>(W2w, W2b16, D_HID * D_HID);

  k_find_last<<<N_NODES / 256, 256, 0, stream>>>(batch, lastI, N_NODES);
  k_vn_q1<<<B_SESS / 8, D_HID, 0, stream>>>(X, lastI, W1w, W1b, v_n, q1s);
  k_alpha<<<N_NODES / 16 / 8, 256, 0, stream>>>(Xb, W2b16, W2b, q1s, batch, qw, qb, alpha);
  k_sg_sh<<<B_SESS, D_HID, 0, stream>>>(X, alpha, lastI, v_n, W3w, W3b, shb);

  dim3 g((V_VOCAB + 63) / 64, B_SESS / 256);
  k_scores<<<g, 256, 0, stream>>>(shb, itemb, out);
}
